// TEMMultiHeadAttention_21242908246652
// MI455X (gfx1250) — matrix operand order repaired and checked by reading, not compiled, not run on hardware
//
#include <hip/hip_runtime.h>
#include <cmath>

typedef __attribute__((ext_vector_type(2)))  float    v2f;
typedef __attribute__((ext_vector_type(8)))  float    v8f;
typedef __attribute__((ext_vector_type(4)))  _Float16 v4h;
typedef __attribute__((ext_vector_type(8)))  _Float16 v8h;
typedef __attribute__((ext_vector_type(16))) _Float16 v16h;

#define DM    1024      // d_model
#define NH    16        // heads
#define DH    64        // head dim
#define LSEQ  2048
#define BN    4
#define MTOK  (BN*LSEQ) // 8192 tokens
#define SROW  2052      // f32 score LDS row stride: 2052%64==4
#define PROW  2056      // f16 prob LDS row stride: 4112B % 256 == 16 -> conflict-free A-frags
#define TROW  1040      // f32 fc-tile LDS row stride

__device__ __forceinline__ v8f vzero8() {
  v8f z = {0.f, 0.f, 0.f, 0.f, 0.f, 0.f, 0.f, 0.f};
  return z;
}

// D(16x16,f32) += A(16x32,f16) * B(32x16,f16)  (codegen-confirmed builtin)
__device__ __forceinline__ v8f wmma16(v16h a, v16h b, v8f c) {
  return __builtin_amdgcn_wmma_f32_16x16x32_f16(false, a, false, b, (short)0, c,
                                                false, false);
}

// A fragment for 16x16x32: lane row = lo; halves 0..15 = row[16*hi+0..15]
// (contiguous k order, matching the contiguous B-fragment loads)
__device__ __forceinline__ v16h loadA16(const _Float16* p, int hi) {
  v8h a0 = *(const v8h*)(p + 16 * hi);
  v8h a1 = *(const v8h*)(p + 16 * hi + 8);
  return __builtin_shufflevector(a0, a1, 0, 1, 2, 3, 4, 5, 6, 7, 8, 9, 10, 11,
                                 12, 13, 14, 15);
}

// ---------------------------------------------------------------------------
// Elementwise f32 -> f16 (vectorized x4)
// ---------------------------------------------------------------------------
__global__ __launch_bounds__(256) void cvt_f32_to_f16(const float* __restrict__ in,
                                                      _Float16* __restrict__ out,
                                                      int n4) {
  int i = blockIdx.x * 256 + threadIdx.x;
  if (i < n4) {
    float4 v = ((const float4*)in)[i];
    v4h h = {(_Float16)v.x, (_Float16)v.y, (_Float16)v.z, (_Float16)v.w};
    ((v4h*)out)[i] = h;
  }
}

// ---------------------------------------------------------------------------
// out[M,N] = A[M,K] @ W[N,K]^T  in f16 (f32 accumulate). One wave -> 16x64.
// transposeOut==1 stores out as [N][M] (xh) so P@V B-frags are contiguous.
// ---------------------------------------------------------------------------
__global__ __launch_bounds__(256) void proj_gemm_f16(const _Float16* __restrict__ A,
                                                     const _Float16* __restrict__ W,
                                                     _Float16* __restrict__ out,
                                                     int transposeOut) {
  const int lane = threadIdx.x & 31;
  const int wave = threadIdx.x >> 5;
  const int gw   = blockIdx.x * 8 + wave;   // 8192 waves
  const int rowTile = gw >> 4;              // 512 row tiles of 16
  const int n0      = (gw & 15) * 64;       // 16 column groups of 64
  const int lo = lane & 15, hi = lane >> 4;

  const _Float16* arow = A + (size_t)(rowTile * 16 + lo) * DM;

  v8f acc0 = vzero8(), acc1 = vzero8(), acc2 = vzero8(), acc3 = vzero8();

  for (int k0 = 0; k0 < DM; k0 += 32) {
    v16h a  = loadA16(arow + k0, hi);
    v16h b0 = *(const v16h*)(W + (size_t)(n0 +  0 + lo) * DM + k0 + 16 * hi);
    v16h b1 = *(const v16h*)(W + (size_t)(n0 + 16 + lo) * DM + k0 + 16 * hi);
    v16h b2 = *(const v16h*)(W + (size_t)(n0 + 32 + lo) * DM + k0 + 16 * hi);
    v16h b3 = *(const v16h*)(W + (size_t)(n0 + 48 + lo) * DM + k0 + 16 * hi);
    acc0 = wmma16(a, b0, acc0);
    acc1 = wmma16(a, b1, acc1);
    acc2 = wmma16(a, b2, acc2);
    acc3 = wmma16(a, b3, acc3);
  }

  v8f accs[4] = {acc0, acc1, acc2, acc3};
  if (transposeOut) {
#pragma unroll
    for (int t = 0; t < 4; ++t)
#pragma unroll
      for (int j = 0; j < 8; ++j)
        out[(size_t)(n0 + t * 16 + lo) * MTOK + rowTile * 16 + j + 8 * hi] =
            (_Float16)accs[t][j];
  } else {
#pragma unroll
    for (int t = 0; t < 4; ++t)
#pragma unroll
      for (int j = 0; j < 8; ++j)
        out[(size_t)(rowTile * 16 + j + 8 * hi) * DM + n0 + t * 16 + lo] =
            (_Float16)accs[t][j];
  }
}

// ---------------------------------------------------------------------------
// One workgroup (8 waves) per (b, h, 16-query tile). Scores accumulate in f32,
// exact softmax over the full row held in LDS, attn written to HBM once,
// probabilities re-quantized to f16 in LDS for the P@V WMMA.
// ---------------------------------------------------------------------------
__global__ __launch_bounds__(256) void attn_kernel(const _Float16* __restrict__ eh,
                                                   const _Float16* __restrict__ xhT,
                                                   float* __restrict__ attn,
                                                   _Float16* __restrict__ featH) {
  extern __shared__ float smem[];
  float* sS    = smem;                 // 16 * SROW f32 scores/probs
  float* sRed  = sS + 16 * SROW;       // 256
  float* sMax  = sRed + 256;           // 16
  float* sSum  = sMax + 16;            // 16
  float* sPart = sSum + 16;            // 4*16*16 split-K partials
  _Float16* sP16 = (_Float16*)(sPart + 4 * 256);  // 16 * PROW f16 probs

  const int tid  = threadIdx.x;
  const int lane = tid & 31;
  const int wave = tid >> 5;
  const int lo = lane & 15, hi = lane >> 4;

  const int qt = blockIdx.x & 127;
  const int h  = (blockIdx.x >> 7) & 15;
  const int b  = blockIdx.x >> 11;

  // ---- Phase 1: S = Q @ K^T (f16 in, f32 out); 1/sqrt(64) applied in f32 ----
  const _Float16* qrow =
      eh + (size_t)(b * LSEQ + qt * 16 + lo) * DM + h * DH;
  v16h qa0 = loadA16(qrow + 0, hi);    // K-block d = 0..31
  v16h qa1 = loadA16(qrow + 32, hi);   // K-block d = 32..63

  for (int kt = wave; kt < 128; kt += 8) {
    const _Float16* krow =
        eh + (size_t)(b * LSEQ + kt * 16 + lo) * DM + h * DH;
    v16h bf0 = *(const v16h*)(krow + 0 + 16 * hi);
    v16h bf1 = *(const v16h*)(krow + 32 + 16 * hi);
    v8f acc = vzero8();
    acc = wmma16(qa0, bf0, acc);
    acc = wmma16(qa1, bf1, acc);
#pragma unroll
    for (int j = 0; j < 8; ++j)
      sS[(j + 8 * hi) * SROW + kt * 16 + lo] = acc[j] * 0.125f;
  }
  __syncthreads();

  // ---- Phase 2: row max, exp, row sum (16 rows x 16 threads) ----
  const int row = tid >> 4;
  const int sub = tid & 15;
  {
    float mx = -INFINITY;
    for (int c = sub; c < LSEQ; c += 16) mx = fmaxf(mx, sS[row * SROW + c]);
    sRed[tid] = mx;
  }
  __syncthreads();
  if (sub == 0) {
    float m = sRed[row * 16];
#pragma unroll
    for (int i = 1; i < 16; ++i) m = fmaxf(m, sRed[row * 16 + i]);
    sMax[row] = m;
  }
  __syncthreads();
  {
    float rm = sMax[row];
    float sm = 0.f;
    for (int c = sub; c < LSEQ; c += 16) {
      float p = __expf(sS[row * SROW + c] - rm);
      sS[row * SROW + c] = p;
      sm += p;
    }
    sRed[tid] = sm;
  }
  __syncthreads();
  if (sub == 0) {
    float s = 0.f;
#pragma unroll
    for (int i = 0; i < 16; ++i) s += sRed[row * 16 + i];
    sSum[row] = s;
  }
  __syncthreads();

  // ---- Phase 3: normalize; write attn (f32, once) + f16 copy for P@V ----
  float* attnOut = attn + (((size_t)(b * NH + h)) * LSEQ + qt * 16) * LSEQ;
  for (int idx = tid; idx < 16 * LSEQ; idx += 256) {
    int r = idx >> 11, c = idx & (LSEQ - 1);
    float p = sS[r * SROW + c] / sSum[r];
    attnOut[(size_t)r * LSEQ + c] = p;
    sP16[r * PROW + c] = (_Float16)p;
  }
  __syncthreads();

  // ---- Phase 4: feat = P @ V, split-K over 2 wave groups, 4 N-tiles ----
  const int ntile = wave & 3;
  const int khalf = wave >> 2;
  const _Float16* vcol =
      xhT + (size_t)(h * DH + ntile * 16 + lo) * MTOK + b * LSEQ;
  const _Float16* prow = sP16 + lo * PROW;
  v8f acc = vzero8();
  const int kbeg = khalf * 1024, kend = kbeg + 1024;
  for (int k0 = kbeg; k0 < kend; k0 += 32) {
    v16h a  = loadA16(prow + k0, hi);                     // LDS, bank-spread
    v16h bf = *(const v16h*)(vcol + k0 + 16 * hi);        // contiguous (xh^T)
    acc = wmma16(a, bf, acc);
  }
  if (khalf == 1) {
#pragma unroll
    for (int j = 0; j < 8; ++j)
      sPart[ntile * 256 + (j + 8 * hi) * 16 + lo] = acc[j];
  }
  __syncthreads();
  if (khalf == 0) {
#pragma unroll
    for (int j = 0; j < 8; ++j) {
      float v = acc[j] + sPart[ntile * 256 + (j + 8 * hi) * 16 + lo];
      featH[(size_t)(b * LSEQ + qt * 16 + j + 8 * hi) * DM + h * DH +
            ntile * 16 + lo] = (_Float16)v;
    }
  }
}

// ---------------------------------------------------------------------------
// out[m,:] = LayerNorm(featH[m,:] @ Wfc^T + x[m,:]) * gamma + beta.
// GEMM in f16/f32-acc; residual + statistics in f32.
// ---------------------------------------------------------------------------
__global__ __launch_bounds__(256) void fc_ln_kernel(const _Float16* __restrict__ featH,
                                                    const _Float16* __restrict__ Wfc,
                                                    const float* __restrict__ xres,
                                                    const float* __restrict__ gamma,
                                                    const float* __restrict__ beta,
                                                    float* __restrict__ out) {
  extern __shared__ float smem[];
  float* sT    = smem;              // 16 * TROW tile (post-residual)
  float* sRedA = sT + 16 * TROW;    // 256
  float* sRedB = sRedA + 256;       // 256
  float* sStat = sRedB + 256;       // 32: mu / rstd

  const int tid  = threadIdx.x;
  const int lane = tid & 31;
  const int wave = tid >> 5;
  const int lo = lane & 15, hi = lane >> 4;
  const int m0 = blockIdx.x * 16;

  const _Float16* arow = featH + (size_t)(m0 + lo) * DM;

  v8f acc[8];
#pragma unroll
  for (int i = 0; i < 8; ++i) acc[i] = vzero8();

  for (int k0 = 0; k0 < DM; k0 += 32) {
    v16h a = loadA16(arow + k0, hi);
#pragma unroll
    for (int i = 0; i < 8; ++i) {
      int n0 = (wave + i * 8) * 16;
      v16h bf = *(const v16h*)(Wfc + (size_t)(n0 + lo) * DM + k0 + 16 * hi);
      acc[i] = wmma16(a, bf, acc[i]);
    }
  }

#pragma unroll
  for (int i = 0; i < 8; ++i) {
    int n0 = (wave + i * 8) * 16;
#pragma unroll
    for (int j = 0; j < 8; ++j) {
      int r = j + 8 * hi, c = n0 + lo;
      sT[r * TROW + c] = acc[i][j] + xres[(size_t)(m0 + r) * DM + c];
    }
  }
  __syncthreads();

  const int row = tid >> 4;
  const int sub = tid & 15;
  {
    float s = 0.f, s2 = 0.f;
    for (int c = sub; c < DM; c += 16) {
      float v = sT[row * TROW + c];
      s += v;
      s2 += v * v;
    }
    sRedA[tid] = s;
    sRedB[tid] = s2;
  }
  __syncthreads();
  if (sub == 0) {
    float s = 0.f, s2 = 0.f;
#pragma unroll
    for (int i = 0; i < 16; ++i) {
      s += sRedA[row * 16 + i];
      s2 += sRedB[row * 16 + i];
    }
    float mu  = s * (1.f / DM);
    float var = s2 * (1.f / DM) - mu * mu;
    sStat[row]      = mu;
    sStat[16 + row] = rsqrtf(var + 1e-6f);
  }
  __syncthreads();

  for (int idx = tid; idx < 16 * DM; idx += 256) {
    int r = idx >> 10, c = idx & (DM - 1);
    float v = (sT[r * TROW + c] - sStat[r]) * sStat[16 + r] * gamma[c] + beta[c];
    out[(size_t)(m0 + r) * DM + c] = v;
  }
}

// ---------------------------------------------------------------------------
extern "C" void kernel_launch(void* const* d_in, const int* in_sizes, int n_in,
                              void* d_out, int out_size, void* d_ws, size_t ws_size,
                              hipStream_t stream) {
  const float* e     = (const float*)d_in[0];
  const float* x     = (const float*)d_in[1];
  const float* We    = (const float*)d_in[2];
  const float* Wx    = (const float*)d_in[3];
  const float* Wfc   = (const float*)d_in[4];
  const float* gamma = (const float*)d_in[5];
  const float* beta  = (const float*)d_in[6];

  float* outF = (float*)d_out;                       // [8192,1024] feat (f32)
  float* attn = outF + (size_t)MTOK * DM;            // [4,16,2048,2048] (f32)

  // f16 workspace layout (~70 MB). featH aliases e_h (dead after proj #1).
  char* ws = (char*)d_ws;
  _Float16* e_h   = (_Float16*)(ws);                          // 16 MB
  _Float16* featH = e_h;                                      // alias
  _Float16* x_h   = (_Float16*)(ws + (size_t)16 * 1024 * 1024);
  _Float16* We_h  = (_Float16*)(ws + (size_t)32 * 1024 * 1024);
  _Float16* Wx_h  = (_Float16*)(ws + (size_t)34 * 1024 * 1024);
  _Float16* Wfc_h = (_Float16*)(ws + (size_t)36 * 1024 * 1024);
  _Float16* eh    = (_Float16*)(ws + (size_t)38 * 1024 * 1024);  // 16 MB
  _Float16* xhT   = (_Float16*)(ws + (size_t)54 * 1024 * 1024);  // 16 MB

  // f32 -> f16 operand conversion (~100 MB total traffic, negligible)
  const int nEX4 = MTOK * DM / 4;     // 2,097,152
  const int nW4  = DM * DM / 4;       //   262,144
  cvt_f32_to_f16<<<nEX4 / 256, 256, 0, stream>>>(e, e_h, nEX4);
  cvt_f32_to_f16<<<nEX4 / 256, 256, 0, stream>>>(x, x_h, nEX4);
  cvt_f32_to_f16<<<nW4 / 256, 256, 0, stream>>>(We, We_h, nW4);
  cvt_f32_to_f16<<<nW4 / 256, 256, 0, stream>>>(Wx, Wx_h, nW4);
  cvt_f32_to_f16<<<nW4 / 256, 256, 0, stream>>>(Wfc, Wfc_h, nW4);

  // Projections (entire intermediate set stays L2-resident)
  proj_gemm_f16<<<1024, 256, 0, stream>>>(e_h, We_h, eh, 0);
  proj_gemm_f16<<<1024, 256, 0, stream>>>(x_h, Wx_h, xhT, 1);

  // Attention: one WG per (b, h, q-tile); full softmax row in 320KB WGP LDS
  size_t shmA = (size_t)(16 * SROW + 256 + 16 + 16 + 4 * 256) * sizeof(float) +
                (size_t)(16 * PROW) * sizeof(_Float16);   // ~198 KB
  attn_kernel<<<BN * NH * (LSEQ / 16), 256, shmA, stream>>>(eh, xhT, attn, featH);

  // Output projection + residual + LayerNorm
  size_t shmC = (size_t)(16 * TROW + 256 + 256 + 32) * sizeof(float);
  fc_ln_kernel<<<MTOK / 16, 256, shmC, stream>>>(featH, Wfc_h, x, gamma, beta, outF);
}
